// ClauseTransformer_59588376264929
// MI455X (gfx1250) — matrix operand order repaired and checked by reading, not compiled, not run on hardware
//
#include <hip/hip_runtime.h>
#include <hip/hip_bf16.h>
#include <math.h>

// Problem constants
#define BB   32
#define SS   1024
#define DD   128
#define HH   4
#define DHH  32
#define LLAY 2
#define NCC  32
#define FFF  512
#define MSROWS (BB * SS)   // 32768 token rows

typedef __bf16 bf16_t;
typedef __attribute__((ext_vector_type(16))) __bf16 v16bf;
typedef __attribute__((ext_vector_type(8)))  float  v8f;

// ---------------------------------------------------------------------------
// WMMA fragment loaders (wave32, 16x16x32 bf16 -> f32)
// A (16x32, MxK): lane L holds row M=L%16; halves of wave split K contiguously:
//   lanes 0-15 : K = 0..15; lanes 16-31 : K = 16..31
// B (32x16, KxN): lane L holds col N=L%16; lanes 0-15 K=0..15, 16-31 K=16..31
// C/D (16x16 f32): VGPR r: lanes 0-15 -> M=r, N=lane; lanes 16-31 -> M=8+r.
// ---------------------------------------------------------------------------
static __device__ inline v16bf load_A_frag(const bf16_t* __restrict__ X, int ld,
                                           int m0, int k0, int lane) {
  const bf16_t* row = X + (size_t)(m0 + (lane & 15)) * ld;
  int kb = k0 + ((lane & 16) ? 16 : 0);
  v16bf a;
#pragma unroll
  for (int g = 0; g < 2; ++g)
#pragma unroll
    for (int w = 0; w < 8; ++w)
      a[g * 8 + w] = row[kb + g * 8 + w];
  return a;
}

// B fragment where memory rows are indexed by N (e.g. torch Linear weight [N,K])
static __device__ inline v16bf load_B_frag_rowN(const bf16_t* __restrict__ W, int ld,
                                                int n0, int k0, int lane) {
  const bf16_t* row = W + (size_t)(n0 + (lane & 15)) * ld;
  int kb = k0 + ((lane & 16) ? 16 : 0);
  v16bf b;
#pragma unroll
  for (int e = 0; e < 16; ++e) b[e] = row[kb + e];
  return b;
}

// B fragment where memory rows are indexed by K (e.g. V matrix [keys, d])
static __device__ inline v16bf load_B_frag_rowK(const bf16_t* __restrict__ V, int ld,
                                                int k0, int n0, int lane) {
  int n  = n0 + (lane & 15);
  int kb = k0 + ((lane & 16) ? 16 : 0);
  v16bf b;
#pragma unroll
  for (int e = 0; e < 16; ++e) b[e] = V[(size_t)(kb + e) * ld + n];
  return b;
}

#define WMMA_BF16(A, Bm, C) \
  __builtin_amdgcn_wmma_f32_16x16x32_bf16(false, (A), false, (Bm), (short)0, (C), false, false)

// ---------------------------------------------------------------------------
// f32 -> bf16 conversion
// ---------------------------------------------------------------------------
__global__ void conv_bf16_kernel(const float* __restrict__ src,
                                 bf16_t* __restrict__ dst, int n) {
  int i = blockIdx.x * blockDim.x + threadIdx.x;
  if (i < n) dst[i] = (bf16_t)src[i];
}

// ---------------------------------------------------------------------------
// Embedding + positional encoding
// ---------------------------------------------------------------------------
__global__ void embed_kernel(const int* __restrict__ tok,
                             const float* __restrict__ emb,
                             const float* __restrict__ pos,
                             float* __restrict__ xf, bf16_t* __restrict__ xb) {
  int bs = blockIdx.x;            // 0 .. B*S-1
  int d  = threadIdx.x;           // 0 .. 127
  int s  = bs % SS;
  int t  = tok[bs];
  float v = emb[(size_t)t * DD + d] + pos[(size_t)s * DD + d];
  xf[(size_t)bs * DD + d] = v;
  xb[(size_t)bs * DD + d] = (bf16_t)v;
}

// ---------------------------------------------------------------------------
// GEMM: C[M,N] = X[M,K] (bf16) * W[N,K]^T (bf16) + bias, optional ReLU.
// One wave computes a 32x64 macro-tile (2 M-tiles x 4 N-tiles -> 8 WMMAs per
// 32-wide K chunk, A fragments reused 4x, B fragments reused 2x).
// grid = (N/64, M/32).
// ---------------------------------------------------------------------------
__global__ void gemm_bf16_kernel(const bf16_t* __restrict__ X,
                                 const bf16_t* __restrict__ W,
                                 const float* __restrict__ bias,
                                 float* __restrict__ Cf, bf16_t* __restrict__ Cb,
                                 int M, int N, int K, int relu) {
  int lane = threadIdx.x;
  int n0 = blockIdx.x * 64;
  int m0 = blockIdx.y * 32;

  v8f acc[2][4];
#pragma unroll
  for (int mi = 0; mi < 2; ++mi)
#pragma unroll
    for (int ni = 0; ni < 4; ++ni) acc[mi][ni] = (v8f){};

  for (int k0 = 0; k0 < K; k0 += 32) {
    // prefetch next K-chunk of the streaming activation tile
    if (k0 + 32 < K)
      __builtin_prefetch(X + (size_t)(m0 + (lane & 15)) * K + k0 + 32, 0, 3);

    v16bf a0 = load_A_frag(X, K, m0, k0, lane);
    v16bf a1 = load_A_frag(X, K, m0 + 16, k0, lane);
    v16bf b0 = load_B_frag_rowN(W, K, n0, k0, lane);
    v16bf b1 = load_B_frag_rowN(W, K, n0 + 16, k0, lane);
    v16bf b2 = load_B_frag_rowN(W, K, n0 + 32, k0, lane);
    v16bf b3 = load_B_frag_rowN(W, K, n0 + 48, k0, lane);

    acc[0][0] = WMMA_BF16(a0, b0, acc[0][0]);
    acc[0][1] = WMMA_BF16(a0, b1, acc[0][1]);
    acc[0][2] = WMMA_BF16(a0, b2, acc[0][2]);
    acc[0][3] = WMMA_BF16(a0, b3, acc[0][3]);
    acc[1][0] = WMMA_BF16(a1, b0, acc[1][0]);
    acc[1][1] = WMMA_BF16(a1, b1, acc[1][1]);
    acc[1][2] = WMMA_BF16(a1, b2, acc[1][2]);
    acc[1][3] = WMMA_BF16(a1, b3, acc[1][3]);
  }

  int nl = lane & 15;
  int rowoff = (lane & 16) ? 8 : 0;
#pragma unroll
  for (int mi = 0; mi < 2; ++mi)
#pragma unroll
    for (int ni = 0; ni < 4; ++ni) {
      int n = n0 + ni * 16 + nl;
      float bv = bias ? bias[n] : 0.0f;
#pragma unroll
      for (int r = 0; r < 8; ++r) {
        int m = m0 + mi * 16 + rowoff + r;
        float v = acc[mi][ni][r] + bv;
        if (relu) v = fmaxf(v, 0.0f);
        size_t idx = (size_t)m * N + n;
        if (Cf) Cf[idx] = v;
        if (Cb) Cb[idx] = (bf16_t)v;
      }
    }
}

// ---------------------------------------------------------------------------
// Flash attention: one wave per (q-tile of 16, head, batch).
// qkv layout [B,S,384]: q = cols h*32.., k = 128+h*32.., v = 256+h*32..
// ---------------------------------------------------------------------------
__global__ void attention_kernel(const bf16_t* __restrict__ qkv,
                                 const int* __restrict__ tok,
                                 bf16_t* __restrict__ o) {
  int lane = threadIdx.x;
  int qt = blockIdx.x, h = blockIdx.y, b = blockIdx.z;
  const float scale = 0.17677669529663687f; // 1/sqrt(32)

  const bf16_t* base = qkv + (size_t)b * SS * (3 * DD);
  const bf16_t* Qb = base + h * DHH;
  const bf16_t* Kb = base + DD + h * DHH;
  const bf16_t* Vb = base + 2 * DD + h * DHH;
  const int* tokb = tok + (size_t)b * SS;

  v16bf aq = load_A_frag(Qb, 3 * DD, qt * 16, 0, lane);

  v8f o0 = {}, o1 = {};
  float m_i[8], l_i[8];
#pragma unroll
  for (int r = 0; r < 8; ++r) { m_i[r] = -1e30f; l_i[r] = 0.0f; }

  __shared__ bf16_t Plds[16 * 32];
  int col = lane & 15;
  int rowoff = (lane & 16) ? 8 : 0;

  for (int j = 0; j < SS / 32; ++j) {
    int kbase = j * 32;
    // scores = Q * K^T (two 16-key tiles)
    v16bf bk0 = load_B_frag_rowN(Kb, 3 * DD, kbase, 0, lane);
    v16bf bk1 = load_B_frag_rowN(Kb, 3 * DD, kbase + 16, 0, lane);
    v8f s0 = {}, s1 = {};
    s0 = WMMA_BF16(aq, bk0, s0);
    s1 = WMMA_BF16(aq, bk1, s1);
    bool pad0 = (tokb[kbase + col] == 0);
    bool pad1 = (tokb[kbase + 16 + col] == 0);

    float p0[8], p1[8];
#pragma unroll
    for (int r = 0; r < 8; ++r) {
      float v0 = pad0 ? -1e9f : s0[r] * scale;
      float v1 = pad1 ? -1e9f : s1[r] * scale;
      float mx = fmaxf(v0, v1);
#pragma unroll
      for (int off = 1; off < 16; off <<= 1) mx = fmaxf(mx, __shfl_xor(mx, off, 32));
      float mn = fmaxf(m_i[r], mx);
      float corr = __expf(m_i[r] - mn);
      float e0 = __expf(v0 - mn);
      float e1 = __expf(v1 - mn);
      float rs = e0 + e1;
#pragma unroll
      for (int off = 1; off < 16; off <<= 1) rs += __shfl_xor(rs, off, 32);
      l_i[r] = l_i[r] * corr + rs;
      m_i[r] = mn;
      o0[r] *= corr;
      o1[r] *= corr;
      p0[r] = e0;
      p1[r] = e1;
    }

    // transpose probabilities (C layout) into A-fragment layout via LDS
    __syncthreads();
#pragma unroll
    for (int r = 0; r < 8; ++r) {
      Plds[(rowoff + r) * 32 + col]      = (bf16_t)p0[r];
      Plds[(rowoff + r) * 32 + 16 + col] = (bf16_t)p1[r];
    }
    __syncthreads();
    v16bf ap;
    {
      int m  = lane & 15;
      int kb = (lane & 16) ? 16 : 0;
#pragma unroll
      for (int g = 0; g < 2; ++g)
#pragma unroll
        for (int w = 0; w < 8; ++w)
          ap[g * 8 + w] = Plds[m * 32 + kb + g * 8 + w];
    }

    // O += P * V   (V chunk is [32 keys x 32 d] -> two B fragments)
    const bf16_t* Vchunk = Vb + (size_t)kbase * (3 * DD);
    v16bf bv0 = load_B_frag_rowK(Vchunk, 3 * DD, 0, 0, lane);
    v16bf bv1 = load_B_frag_rowK(Vchunk, 3 * DD, 0, 16, lane);
    o0 = WMMA_BF16(ap, bv0, o0);
    o1 = WMMA_BF16(ap, bv1, o1);
  }

#pragma unroll
  for (int r = 0; r < 8; ++r) {
    float inv = 1.0f / fmaxf(l_i[r], 1e-20f);
    int sg = qt * 16 + rowoff + r;
    size_t idx = ((size_t)b * SS + sg) * DD + h * DHH;
    o[idx + col]      = (bf16_t)(o0[r] * inv);
    o[idx + 16 + col] = (bf16_t)(o1[r] * inv);
  }
}

// ---------------------------------------------------------------------------
// x = LayerNorm(xin + add) * w + b ; writes f32 and bf16 mirrors.
// One block (128 threads) per token row.
// ---------------------------------------------------------------------------
__global__ void add_ln_kernel(const float* __restrict__ xin,
                              const float* __restrict__ add,
                              const float* __restrict__ w,
                              const float* __restrict__ b,
                              float* __restrict__ xf, bf16_t* __restrict__ xb) {
  __shared__ float red[4];
  int row = blockIdx.x;
  int d = threadIdx.x;
  size_t idx = (size_t)row * DD + d;
  float v = xin[idx] + add[idx];

  float s = v;
  for (int off = 16; off > 0; off >>= 1) s += __shfl_xor(s, off, 32);
  if ((d & 31) == 0) red[d >> 5] = s;
  __syncthreads();
  float mean = (red[0] + red[1] + red[2] + red[3]) * (1.0f / DD);
  __syncthreads();
  float c = v - mean;
  float s2 = c * c;
  for (int off = 16; off > 0; off >>= 1) s2 += __shfl_xor(s2, off, 32);
  if ((d & 31) == 0) red[d >> 5] = s2;
  __syncthreads();
  float var = (red[0] + red[1] + red[2] + red[3]) * (1.0f / DD);

  float y = c * rsqrtf(var + 1e-5f) * w[d] + b[d];
  xf[idx] = y;
  xb[idx] = (bf16_t)y;
}

// ---------------------------------------------------------------------------
// Clause pooling + clause MLP + mean over non-empty + LN + head.
// One block (128 threads) per batch element.
// ---------------------------------------------------------------------------
static __device__ inline float gelu_exact(float x) {
  return 0.5f * x * (1.0f + erff(x * 0.70710678118f));
}

__global__ void pool_head_kernel(const float* __restrict__ x,
                                 const int* __restrict__ cid_all,
                                 const float* __restrict__ cw, const float* __restrict__ cb,
                                 const float* __restrict__ nw, const float* __restrict__ nb,
                                 const float* __restrict__ c1w, const float* __restrict__ c1b,
                                 const float* __restrict__ c2w, const float* __restrict__ c2b,
                                 float* __restrict__ out) {
  int b = blockIdx.x, t = threadIdx.x;
  __shared__ float ssum[NCC][DD];
  __shared__ float cnt[NCC];
  __shared__ float vecA[DD];
  __shared__ float hh[DD / 2];
  __shared__ float red[4];

  const int* cid = cid_all + (size_t)b * SS;
  const float* xb = x + (size_t)b * SS * DD;

  for (int c = 0; c < NCC; ++c) ssum[c][t] = 0.0f;
  if (t < NCC) {
    int c = 0;
    for (int s = 0; s < SS; ++s) c += (cid[s] == t);
    cnt[t] = (float)c;
  }
  __syncthreads();

  // segment sums: thread t owns feature column t -> no atomics needed
  for (int s = 0; s < SS; ++s) ssum[cid[s]][t] += xb[(size_t)s * DD + t];
  __syncthreads();

  for (int c = 0; c < NCC; ++c) ssum[c][t] /= fmaxf(cnt[c], 1.0f);
  __syncthreads();

  // clause MLP (output feature = t), mean over non-empty clauses
  float acc = 0.0f, ne = 0.0f;
  for (int c = 0; c < NCC; ++c) {
    float dot = cb[t];
    for (int d = 0; d < DD; ++d) dot += ssum[c][d] * cw[(size_t)t * DD + d];
    float g = gelu_exact(dot);
    if (cnt[c] > 0.0f) { acc += g; ne += 1.0f; }
  }
  float pooled = acc / fmaxf(ne, 1.0f);

  // LayerNorm over pooled
  float s = pooled;
  for (int off = 16; off > 0; off >>= 1) s += __shfl_xor(s, off, 32);
  if ((t & 31) == 0) red[t >> 5] = s;
  __syncthreads();
  float mean = (red[0] + red[1] + red[2] + red[3]) * (1.0f / DD);
  __syncthreads();
  float c0 = pooled - mean;
  float s2 = c0 * c0;
  for (int off = 16; off > 0; off >>= 1) s2 += __shfl_xor(s2, off, 32);
  if ((t & 31) == 0) red[t >> 5] = s2;
  __syncthreads();
  float var = (red[0] + red[1] + red[2] + red[3]) * (1.0f / DD);
  vecA[t] = c0 * rsqrtf(var + 1e-5f) * nw[t] + nb[t];
  __syncthreads();

  if (t < DD / 2) {
    float dot = c1b[t];
    for (int d = 0; d < DD; ++d) dot += vecA[d] * c1w[(size_t)t * DD + d];
    hh[t] = gelu_exact(dot);
  }
  __syncthreads();
  if (t == 0) {
    float sfin = c2b[0];
    for (int oo = 0; oo < DD / 2; ++oo) sfin += hh[oo] * c2w[oo];
    out[b] = sfin;
  }
}

// ---------------------------------------------------------------------------
// Launch
// ---------------------------------------------------------------------------
extern "C" void kernel_launch(void* const* d_in, const int* in_sizes, int n_in,
                              void* d_out, int out_size, void* d_ws, size_t ws_size,
                              hipStream_t stream) {
  (void)in_sizes; (void)n_in; (void)out_size; (void)ws_size;

  const int*   tok    = (const int*)d_in[0];
  const int*   cid    = (const int*)d_in[1];
  const float* emb    = (const float*)d_in[2];
  const float* pos    = (const float*)d_in[3];
  const float* in_w   = (const float*)d_in[4];
  const float* in_b   = (const float*)d_in[5];
  const float* out_w  = (const float*)d_in[6];
  const float* out_b  = (const float*)d_in[7];
  const float* lin1_w = (const float*)d_in[8];
  const float* lin1_b = (const float*)d_in[9];
  const float* lin2_w = (const float*)d_in[10];
  const float* lin2_b = (const float*)d_in[11];
  const float* ln1_w  = (const float*)d_in[12];
  const float* ln1_b  = (const float*)d_in[13];
  const float* ln2_w  = (const float*)d_in[14];
  const float* ln2_b  = (const float*)d_in[15];
  const float* cw     = (const float*)d_in[16];
  const float* cbb    = (const float*)d_in[17];
  const float* nw     = (const float*)d_in[18];
  const float* nb     = (const float*)d_in[19];
  const float* c1w    = (const float*)d_in[20];
  const float* c1b    = (const float*)d_in[21];
  const float* c2w    = (const float*)d_in[22];
  const float* c2b    = (const float*)d_in[23];
  float* out = (float*)d_out;

  // workspace carve-out
  size_t off = 0;
  auto wsalloc = [&](size_t bytes) -> void* {
    void* p = (char*)d_ws + off;
    off = (off + bytes + 255) & ~(size_t)255;
    return p;
  };
  float*  x_f   = (float*)wsalloc((size_t)MSROWS * DD * 4);
  bf16_t* x_b   = (bf16_t*)wsalloc((size_t)MSROWS * DD * 2);
  bf16_t* qkv_b = (bf16_t*)wsalloc((size_t)MSROWS * 3 * DD * 2);
  bf16_t* o_b   = (bf16_t*)wsalloc((size_t)MSROWS * DD * 2);
  float*  tmp_f = (float*)wsalloc((size_t)MSROWS * DD * 4);
  bf16_t* h1_b  = (bf16_t*)wsalloc((size_t)MSROWS * FFF * 2);
  bf16_t* in_w_b   = (bf16_t*)wsalloc((size_t)LLAY * 3 * DD * DD * 2);
  bf16_t* out_w_b  = (bf16_t*)wsalloc((size_t)LLAY * DD * DD * 2);
  bf16_t* lin1_w_b = (bf16_t*)wsalloc((size_t)LLAY * FFF * DD * 2);
  bf16_t* lin2_w_b = (bf16_t*)wsalloc((size_t)LLAY * DD * FFF * 2);

  auto conv = [&](const float* s, bf16_t* d, int n) {
    conv_bf16_kernel<<<(n + 255) / 256, 256, 0, stream>>>(s, d, n);
  };
  conv(in_w, in_w_b, LLAY * 3 * DD * DD);
  conv(out_w, out_w_b, LLAY * DD * DD);
  conv(lin1_w, lin1_w_b, LLAY * FFF * DD);
  conv(lin2_w, lin2_w_b, LLAY * DD * FFF);

  embed_kernel<<<MSROWS, DD, 0, stream>>>(tok, emb, pos, x_f, x_b);

  for (int l = 0; l < LLAY; ++l) {
    // qkv = x @ in_w[l]^T + in_b[l]     [32768, 384]
    gemm_bf16_kernel<<<dim3((3 * DD) / 64, MSROWS / 32), 32, 0, stream>>>(
        x_b, in_w_b + (size_t)l * 3 * DD * DD, in_b + (size_t)l * 3 * DD,
        nullptr, qkv_b, MSROWS, 3 * DD, DD, 0);

    // attention -> o_b [32768, 128]
    attention_kernel<<<dim3(SS / 16, HH, BB), 32, 0, stream>>>(qkv_b, tok, o_b);

    // out projection -> tmp_f (f32 for residual)
    gemm_bf16_kernel<<<dim3(DD / 64, MSROWS / 32), 32, 0, stream>>>(
        o_b, out_w_b + (size_t)l * DD * DD, out_b + (size_t)l * DD,
        tmp_f, nullptr, MSROWS, DD, DD, 0);

    // x = LN(x + o)
    add_ln_kernel<<<MSROWS, DD, 0, stream>>>(x_f, tmp_f,
        ln1_w + (size_t)l * DD, ln1_b + (size_t)l * DD, x_f, x_b);

    // h1 = relu(x @ lin1^T + b)  [32768, 512]
    gemm_bf16_kernel<<<dim3(FFF / 64, MSROWS / 32), 32, 0, stream>>>(
        x_b, lin1_w_b + (size_t)l * FFF * DD, lin1_b + (size_t)l * FFF,
        nullptr, h1_b, MSROWS, FFF, DD, 1);

    // f = h1 @ lin2^T + b  -> tmp_f
    gemm_bf16_kernel<<<dim3(DD / 64, MSROWS / 32), 32, 0, stream>>>(
        h1_b, lin2_w_b + (size_t)l * DD * FFF, lin2_b + (size_t)l * DD,
        tmp_f, nullptr, MSROWS, DD, FFF, 0);

    // x = LN(x + f)
    add_ln_kernel<<<MSROWS, DD, 0, stream>>>(x_f, tmp_f,
        ln2_w + (size_t)l * DD, ln2_b + (size_t)l * DD, x_f, x_b);
  }

  pool_head_kernel<<<BB, DD, 0, stream>>>(x_f, cid, cw, cbb, nw, nb,
                                          c1w, c1b, c2w, c2b, out);
}
